// SparseSubsampledTriangleAttention_7627861918047
// MI455X (gfx1250) — matrix operand order repaired and checked by reading, not compiled, not run on hardware
//
#include <hip/hip_runtime.h>
#include <hip/hip_bf16.h>

// ---------------------------------------------------------------------------
// SparseSubsampledTriangleAttention for MI455X (gfx1250, wave32, WMMA + TDM)
//
// N=384 nodes, K=60 nbrs, SUB=20, C_Z=128, H=4, D=32, E=23040 edges.
//
//  * gate + rbf-bias collapse to a 384x384x4 pairwise table (2.4 MB, L2).
//  * big GEMMs (edges@[Wqk|Wv]: 23040x128x384, upd@Wout: 23040x128x128) run
//    on v_wmma_f32_16x16x32_bf16; weights staged into LDS once per block via
//    tensor_load_to_lds (TDM) + s_wait_tensorcnt, read back with ds_load.
//  * each wave computes TWO 16-row tiles with alternating accumulators so
//    consecutive WMMAs are independent (hides D->C hazard NOPs) and every
//    LDS B-fetch feeds 2x the math.
// ---------------------------------------------------------------------------

#define N_NODES 384
#define KNBR    60
#define SUBK    20
#define NHEAD   4
#define DHEAD   32
#define NE      (N_NODES * KNBR)   // 23040
#define NRBF    64

typedef __attribute__((ext_vector_type(16))) __bf16 v16bf;
typedef __attribute__((ext_vector_type(8)))  __bf16 v8bf;
typedef __attribute__((ext_vector_type(8)))  float  v8f;
typedef __attribute__((ext_vector_type(4)))  unsigned int u32x4;
typedef __attribute__((ext_vector_type(8)))  int i32x8;
typedef __attribute__((ext_vector_type(4)))  int i32x4;

__device__ __forceinline__ unsigned short f2bf(float f) {
    union { float f; unsigned int u; } v; v.f = f;
    unsigned int u = v.u;
    unsigned int r = u + 0x7FFFu + ((u >> 16) & 1u);   // round-to-nearest-even
    return (unsigned short)(r >> 16);
}

// --------------------------------------------------------------------------
// g1[v,h] = nf[v]·W_gate[0:128,h] ; g2[v,h] = nf[v]·W_gate[128:256,h]
__global__ __launch_bounds__(256) void gate_proj_kernel(
    const float* __restrict__ nf, const float* __restrict__ W_gate,
    float* __restrict__ g1, float* __restrict__ g2)
{
    int t = blockIdx.x * 256 + threadIdx.x;           // < 3072
    int side = (t >= 1536) ? 1 : 0;
    int u = t - side * 1536;
    int node = u >> 2, h = u & 3;
    const float* row = nf + node * 128;
    const float* wg  = W_gate + side * 128 * 4 + h;
    float acc = 0.f;
    #pragma unroll 8
    for (int k = 0; k < 128; ++k) acc += row[k] * wg[k * 4];
    (side ? g2 : g1)[node * 4 + h] = acc;
}

// --------------------------------------------------------------------------
// Btab[a,b,h] = sigmoid(g1[a,h]+g2[b,h]+bg[h]) * (rbf(dist(a,b))@W_db + bdb)[h]
__global__ __launch_bounds__(256) void pair_bias_kernel(
    const float* __restrict__ trans,
    const float* __restrict__ g1, const float* __restrict__ g2,
    const float* __restrict__ W_db, const float* __restrict__ b_db,
    const float* __restrict__ b_gate, float* __restrict__ Btab)
{
    int t = blockIdx.x * 256 + threadIdx.x;           // < 147456
    int a = t / N_NODES, b = t - a * N_NODES;
    float dx = trans[a * 3 + 0] - trans[b * 3 + 0] + 1e-8f;
    float dy = trans[a * 3 + 1] - trans[b * 3 + 1] + 1e-8f;
    float dz = trans[a * 3 + 2] - trans[b * 3 + 2] + 1e-8f;
    float d  = sqrtf(dx * dx + dy * dy + dz * dz);

    float acc0 = b_db[0], acc1 = b_db[1], acc2 = b_db[2], acc3 = b_db[3];
    const float step = 20.0f / 63.0f;
    const float inv_sigma = 64.0f / 20.0f;            // 1/sigma = 3.2
    #pragma unroll 4
    for (int i = 0; i < NRBF; ++i) {
        float x = (d - step * (float)i) * inv_sigma;
        float r = __expf(-x * x);
        acc0 += r * W_db[i * 4 + 0];
        acc1 += r * W_db[i * 4 + 1];
        acc2 += r * W_db[i * 4 + 2];
        acc3 += r * W_db[i * 4 + 3];
    }
    float* o = Btab + (size_t)t * 4;
    float db[4] = {acc0, acc1, acc2, acc3};
    #pragma unroll
    for (int h = 0; h < 4; ++h) {
        float z = g1[a * 4 + h] + g2[b * 4 + h] + b_gate[h];
        float g = 1.0f / (1.0f + __expf(-z));
        o[h] = g * db[h];
    }
}

// --------------------------------------------------------------------------
__global__ __launch_bounds__(256) void to_bf16_kernel(
    const float* __restrict__ in, unsigned short* __restrict__ out, int n)
{
    int t = blockIdx.x * 256 + threadIdx.x;
    if (t < n) out[t] = f2bf(in[t]);
}

// --------------------------------------------------------------------------
// Pre-pack weights into WMMA B-operand per-lane layout:
// Wpack[((nt*4 + c)*32 + lane)*16 + i] = W[c*32 + (lane>>4)*8 + i + (i>=8)*8,
//                                         nt*16 + (lane&15)]
__global__ __launch_bounds__(256) void pack_w_kernel(
    const float* __restrict__ W0, int ld0,
    const float* __restrict__ W1, int ld1, int split,
    int ntiles, unsigned short* __restrict__ Wpack)
{
    int t = blockIdx.x * 256 + threadIdx.x;
    int nt = t >> 7;
    if (nt >= ntiles) return;
    int lane = t & 31;
    int c    = (t >> 5) & 3;
    int col  = nt * 16 + (lane & 15);
    int kbase = c * 32 + (lane >> 4) * 8;
    unsigned short* o = Wpack + (size_t)t * 16;
    #pragma unroll
    for (int i = 0; i < 16; ++i) {
        int k = kbase + i + (i >= 8 ? 8 : 0);
        float v = (col < split) ? W0[k * ld0 + col] : W1[k * ld1 + (col - split)];
        o[i] = f2bf(v);
    }
}

// --------------------------------------------------------------------------
// TDM: DMA nbytes from global to LDS offset 0 (1 row of 8-byte elements).
// D# per CDNA5 ISA ch.8; 6-arg builtin arity (clang-23 toolchain).
__device__ __forceinline__ void tdm_load_to_lds(const void* gsrc,
                                                unsigned int nbytes)
{
    unsigned long long ga = (unsigned long long)(size_t)gsrc;
    unsigned int n8 = nbytes >> 3;                       // 8-byte elements
    u32x4 g0;
    g0[0] = 1u;                                          // count=1, user mode
    g0[1] = 0u;                                          // lds_addr = 0
    g0[2] = (unsigned int)(ga & 0xFFFFFFFFu);            // global_addr[95:64]
    g0[3] = (unsigned int)((ga >> 32) & 0x01FFFFFFu)     // global_addr msbs
          | (2u << 30);                                  // type = 2 (image)
    i32x8 g1;
    g1[0] = (int)(3u << 16);                             // data_size = 8B
    g1[1] = (int)((n8 & 0xFFFFu) << 16);                 // tensor_dim0 lo16
    g1[2] = (int)(((n8 >> 16) & 0xFFFFu) | (1u << 16));  // dim0 hi | dim1=1
    g1[3] = (int)((n8 & 0xFFFFu) << 16);                 // tile_dim0 = n8
    g1[4] = 1;                                           // tile_dim1=1, dim2=0
    g1[5] = (int)n8;                                     // dim0_stride lo32
    g1[6] = 0;
    g1[7] = 0;
    i32x4 gz4 = {0, 0, 0, 0};
    i32x8 gz8 = {0, 0, 0, 0, 0, 0, 0, 0};
    __builtin_amdgcn_tensor_load_to_lds(g0, g1, gz4, gz4, gz8, 0);
}

// --------------------------------------------------------------------------
// WMMA GEMM:  Out[M=NE, ncols] = bf16(A[M,128]) @ Wpack + bias
// Block: 8 waves, each wave computes TWO 16-row tiles (alternating
// accumulators). Weights live in LDS (TDM-staged), read via ds_load.
__global__ __launch_bounds__(256) void gemm_wmma_kernel(
    const unsigned short* __restrict__ Abf,     // [M,128] bf16 row-major
    const unsigned short* __restrict__ Wpack,   // packed B operands (global)
    const float* __restrict__ bias0, const float* __restrict__ bias1, int split,
    int ntiles, float* __restrict__ Out, int ldo)
{
    __shared__ unsigned short lds_w[49152];     // 96 KB, fits QKV pack

    // ---- stage weights into LDS with the Tensor Data Mover (wave 0 only)
    if (threadIdx.x < 32) {
        tdm_load_to_lds(Wpack, (unsigned int)(ntiles * 4 * 32 * 16 * 2));
        __builtin_amdgcn_s_wait_tensorcnt(0);
    }
    __syncthreads();
    // The TDM builtin carries no pointer to lds_w, so without this the
    // compiler treats lds_w as never-written and folds the ds_loads to undef.
    // Escaping the address through an asm with a memory clobber makes the
    // TDM's LDS writes visible to the optimizer.
    {
        unsigned short* p = &lds_w[0];
        asm volatile("" : "+v"(p) : : "memory");
    }

    int wid  = threadIdx.x >> 5;
    int lane = threadIdx.x & 31;
    int half = lane >> 4;
    int lr   = lane & 15;
    int mt   = blockIdx.x * 16 + wid * 2;       // first of two m-tiles
    int m0   = mt * 16;
    int m1   = m0 + 16;

    // A operand: lane l<16 -> row l, K {c*32+0..7, c*32+16..23};
    //            lane l+16 -> row l, K {c*32+8..15, c*32+24..31}.
    const unsigned short* ar0 = Abf + (size_t)(m0 + lr) * 128;
    const unsigned short* ar1 = Abf + (size_t)(m1 + lr) * 128;
    v16bf a0[4], a1[4];
    #pragma unroll
    for (int c = 0; c < 4; ++c) {
        v8bf lo0 = *reinterpret_cast<const v8bf*>(ar0 + c * 32 +      half * 8);
        v8bf hi0 = *reinterpret_cast<const v8bf*>(ar0 + c * 32 + 16 + half * 8);
        v8bf lo1 = *reinterpret_cast<const v8bf*>(ar1 + c * 32 +      half * 8);
        v8bf hi1 = *reinterpret_cast<const v8bf*>(ar1 + c * 32 + 16 + half * 8);
        #pragma unroll
        for (int i = 0; i < 8; ++i) {
            a0[c][i] = lo0[i]; a0[c][8 + i] = hi0[i];
            a1[c][i] = lo1[i]; a1[c][8 + i] = hi1[i];
        }
    }

    for (int nt = 0; nt < ntiles; ++nt) {
        v8f acc0 = {}, acc1 = {};
        #pragma unroll
        for (int c = 0; c < 4; ++c) {
            const unsigned short* wp = lds_w + ((nt * 4 + c) * 32 + lane) * 16;
            v8bf lo = *reinterpret_cast<const v8bf*>(wp);
            v8bf hi = *reinterpret_cast<const v8bf*>(wp + 8);
            v16bf b;
            #pragma unroll
            for (int i = 0; i < 8; ++i) { b[i] = lo[i]; b[8 + i] = hi[i]; }
            acc0 = __builtin_amdgcn_wmma_f32_16x16x32_bf16(
                false, a0[c], false, b, (short)0, acc0, false, false);
            acc1 = __builtin_amdgcn_wmma_f32_16x16x32_bf16(
                false, a1[c], false, b, (short)0, acc1, false, false);
        }
        // C/D layout: lane&15 = column, VGPR r = row r + 8*(lane>=16)
        int col = nt * 16 + lr;
        float bv = (col < split) ? bias0[col] : bias1[col - split];
        #pragma unroll
        for (int r = 0; r < 8; ++r) {
            int row = m0 + r + 8 * half;
            Out[(size_t)row * ldo + col] = acc0[r] + bv;
        }
        #pragma unroll
        for (int r = 0; r < 8; ++r) {
            int row = m1 + r + 8 * half;
            Out[(size_t)row * ldo + col] = acc1[r] + bv;
        }
    }
}

// --------------------------------------------------------------------------
// Attention core: one thread per (edge, head).  QKV row layout:
//   [0:128)=q, [128:256)=k, [256:384)=v  (head h occupies h*32..h*32+31)
__global__ __launch_bounds__(256) void attn_kernel(
    const float* __restrict__ QKV, const float* __restrict__ Btab,
    const int* __restrict__ edge_index, const int* __restrict__ subidx,
    unsigned short* __restrict__ updbf)
{
    int t = blockIdx.x * 256 + threadIdx.x;   // < NE*4 exactly
    int e = t >> 2, h = t & 3;
    int n = e / KNBR;
    int a_node = edge_index[e];               // dst row of edge_index[0]

    const float* qrow = QKV + (size_t)e * 384 + h * DHEAD;
    float q[DHEAD];
    #pragma unroll
    for (int d = 0; d < DHEAD; ++d) q[d] = qrow[d];

    float logit[SUBK];
    int   e2s[SUBK];
    float mx = -1e30f;
    const float scale = 0.08838834764831845f;  // 1/sqrt(128)
    #pragma unroll 2
    for (int s = 0; s < SUBK; ++s) {
        int idx = subidx[e * SUBK + s];
        int e2  = n * KNBR + idx;
        e2s[s] = e2;
        const float* krow = QKV + (size_t)e2 * 384 + 128 + h * DHEAD;
        float dot = 0.f;
        #pragma unroll
        for (int d = 0; d < DHEAD; ++d) dot += q[d] * krow[d];
        int b_node = edge_index[e2];
        float lg = dot * scale + Btab[((size_t)a_node * N_NODES + b_node) * 4 + h];
        logit[s] = lg;
        mx = fmaxf(mx, lg);
    }
    float sum = 0.f;
    #pragma unroll
    for (int s = 0; s < SUBK; ++s) { float p = __expf(logit[s] - mx); logit[s] = p; sum += p; }
    float inv = 1.0f / sum;

    float upd[DHEAD];
    #pragma unroll
    for (int d = 0; d < DHEAD; ++d) upd[d] = 0.f;
    #pragma unroll 2
    for (int s = 0; s < SUBK; ++s) {
        const float* vrow = QKV + (size_t)e2s[s] * 384 + 256 + h * DHEAD;
        float p = logit[s] * inv;
        #pragma unroll
        for (int d = 0; d < DHEAD; ++d) upd[d] += p * vrow[d];
    }
    unsigned short* o = updbf + (size_t)e * 128 + h * DHEAD;
    #pragma unroll
    for (int d = 0; d < DHEAD; ++d) o[d] = f2bf(upd[d]);
}

// --------------------------------------------------------------------------
extern "C" void kernel_launch(void* const* d_in, const int* in_sizes, int n_in,
                              void* d_out, int out_size, void* d_ws, size_t ws_size,
                              hipStream_t stream)
{
    (void)in_sizes; (void)n_in; (void)out_size; (void)ws_size;

    const float* node_features = (const float*)d_in[0];
    const float* node_trans    = (const float*)d_in[1];
    const float* edge_features = (const float*)d_in[2];
    const int*   edge_index    = (const int*)  d_in[3];
    const int*   subsample_idx = (const int*)  d_in[4];
    const float* W_gate        = (const float*)d_in[5];
    const float* b_gate        = (const float*)d_in[6];
    const float* W_db          = (const float*)d_in[7];
    const float* b_db          = (const float*)d_in[8];
    const float* W_qk          = (const float*)d_in[9];
    const float* b_qk          = (const float*)d_in[10];
    const float* W_v           = (const float*)d_in[11];
    const float* b_v           = (const float*)d_in[12];
    const float* W_out         = (const float*)d_in[13];
    const float* b_out         = (const float*)d_in[14];
    float* out = (float*)d_out;

    // workspace layout (bytes, all >=512B aligned)
    char* ws = (char*)d_ws;
    float*          g1    = (float*)         (ws + 0);          //   6,144
    float*          g2    = (float*)         (ws + 6144);       //   6,144
    float*          Btab  = (float*)         (ws + 12288);      // 2,359,296
    unsigned short* Abf   = (unsigned short*)(ws + 2371584);    // 5,898,240
    unsigned short* Wqkvp = (unsigned short*)(ws + 8269824);    //    98,304
    unsigned short* Woutp = (unsigned short*)(ws + 8368128);    //    32,768
    float*          QKV   = (float*)         (ws + 8400896);    // 35,389,440
    unsigned short* updbf = (unsigned short*)(ws + 43790336);   // 5,898,240
                                                                // = 49,688,576

    // 1) per-node gate projections (3072 threads)
    gate_proj_kernel<<<12, 256, 0, stream>>>(node_features, W_gate, g1, g2);

    // 2) edges f32 -> bf16 (E*128 = 2,949,120 elements)
    to_bf16_kernel<<<11520, 256, 0, stream>>>(edge_features, Abf, NE * 128);

    // 3) pack weights into WMMA B layout: [Wqk | Wv] (24 col tiles), Wout (8)
    pack_w_kernel<<<12, 256, 0, stream>>>(W_qk, 256, W_v, 128, 256, 24, Wqkvp);
    pack_w_kernel<<<4,  256, 0, stream>>>(W_out, 128, W_out, 128, 128, 8, Woutp);

    // 4) pairwise gate*rbf-bias table (384*384 threads)
    pair_bias_kernel<<<576, 256, 0, stream>>>(node_trans, g1, g2, W_db, b_db,
                                              b_gate, Btab);

    // 5) QKV GEMM: [23040,128]@[128,384], TDM-staged weights, 16 m-tiles/block
    gemm_wmma_kernel<<<90, 256, 0, stream>>>(Abf, Wqkvp, b_qk, b_v, 256,
                                             24, QKV, 384);

    // 6) attention core (E*4 = 92160 threads)
    attn_kernel<<<360, 256, 0, stream>>>(QKV, Btab, edge_index, subsample_idx,
                                         updbf);

    // 7) output GEMM: [23040,128]@[128,128] + b_out
    gemm_wmma_kernel<<<90, 256, 0, stream>>>(updbf, Woutp, b_out, b_out, 128,
                                             8, out, 128);
}
